// MyMultiheadAttention_64226940944669
// MI455X (gfx1250) — matrix operand order repaired and checked by reading, not compiled, not run on hardware
//
#include <hip/hip_runtime.h>

// ---------------- CDNA5 / gfx1250 multi-head attention ----------------
// Pipeline: weight transpose->bf16, QKV projections (WMMA bf16, async-LDS
// double-buffered weight tiles), flash-attention (WMMA bf16, async-LDS
// double-buffered K/V tiles, online softmax, LDS P-transpose), output
// projection (WMMA bf16, f32 out).

typedef __attribute__((ext_vector_type(16))) __bf16 v16bf;
typedef __attribute__((ext_vector_type(8)))  __bf16 v8bf;
typedef __attribute__((ext_vector_type(8)))  float  v8f;

constexpr int S_LEN = 2048;
constexpr int EMB   = 1024;
constexpr int NH    = 16;
constexpr int HD    = 64;
constexpr int BATCH = 2;
constexpr int MROWS = BATCH * S_LEN;   // 4096

// ---- gfx1250 async global->LDS copy (16 bytes per lane) ---------------
// GLOBAL_LOAD_ASYNC_TO_LDS_B128: VDST = per-lane LDS byte address,
// VADDR = per-lane 64-bit global address (GV mode). Tracked on ASYNCcnt.
__device__ __forceinline__ void async_copy16(const void* gptr, void* lptr) {
  unsigned lds = (unsigned)(uintptr_t)lptr;   // low 32 bits = LDS offset
  unsigned long long ga = (unsigned long long)(uintptr_t)gptr;
  asm volatile("global_load_async_to_lds_b128 %0, %1, off"
               :: "v"(lds), "v"(ga) : "memory");
}
__device__ __forceinline__ void wait_async_le(int n) {
  if (n == 0)      asm volatile("s_wait_asynccnt 0" ::: "memory");
  else if (n == 1) asm volatile("s_wait_asynccnt 1" ::: "memory");
  else             asm volatile("s_wait_asynccnt 2" ::: "memory");
}

// ---- fragment helpers -------------------------------------------------
// A-matrix 16x32 bf16 k order used here (contiguous, B matches): lane holds row M=lane&15;
// elements 0..7  <-> K = half*16 + j      (half = lane>>4)
// elements 8..15 <-> K = half*16 + 8 + j

__device__ __forceinline__ v16bf cvt_a_from_f32(float4 f0, float4 f1,
                                                float4 f2, float4 f3) {
  v16bf a;
  a[0]=(__bf16)f0.x;  a[1]=(__bf16)f0.y;  a[2]=(__bf16)f0.z;  a[3]=(__bf16)f0.w;
  a[4]=(__bf16)f1.x;  a[5]=(__bf16)f1.y;  a[6]=(__bf16)f1.z;  a[7]=(__bf16)f1.w;
  a[8]=(__bf16)f2.x;  a[9]=(__bf16)f2.y;  a[10]=(__bf16)f2.z; a[11]=(__bf16)f2.w;
  a[12]=(__bf16)f3.x; a[13]=(__bf16)f3.y; a[14]=(__bf16)f3.z; a[15]=(__bf16)f3.w;
  return a;
}

__device__ __forceinline__ v16bf load_a_from_bf16(const __bf16* p1, const __bf16* p2) {
  v8bf r1 = *(const v8bf*)p1;
  v8bf r2 = *(const v8bf*)p2;
  v16bf a;
#pragma unroll
  for (int j = 0; j < 8; ++j) { a[j] = r1[j]; a[j + 8] = r2[j]; }
  return a;
}

__device__ __forceinline__ v8f wmma_bf16(v16bf a, v16bf b, v8f c) {
  return __builtin_amdgcn_wmma_f32_16x16x32_bf16(false, a, false, b, (short)0, c,
                                                 false, false);
}

// ---- kernel 1: weight transpose + f32->bf16 ---------------------------
__global__ __launch_bounds__(256) void wtrans_kernel(const float* __restrict__ W,
                                                     __bf16* __restrict__ WT) {
  const size_t id = (size_t)blockIdx.x * 256 + threadIdx.x;   // 0 .. EMB*EMB-1
  const int kk = (int)(id & (EMB - 1));
  const int n  = (int)(id >> 10);
  WT[(size_t)n * EMB + kk] = (__bf16)W[(size_t)kk * EMB + n];
}

// ---- kernel 2: input projections  Y = X @ W  --------------------------
// Block = 8 waves sharing one 64-column weight tile (async-staged in LDS,
// double buffered); each wave owns a distinct 16-row tile -> 16x64 output.
// mode 0: store bf16 [B,H,S,D] (q,k); mode 2: store bf16 [B,H,D,S] (v^T).
__global__ __launch_bounds__(256) void proj_in_kernel(const float* __restrict__ X,
                                                      const __bf16* __restrict__ WT,
                                                      __bf16* __restrict__ out,
                                                      int mode) {
  __shared__ __align__(32) __bf16 bbufs[2][64 * 32];   // [n_local][k_local]

  const int tid  = threadIdx.x;
  const int lane = tid & 31, wib = tid >> 5;
  const int half = lane >> 4, l16 = lane & 15;
  const int colTile  = blockIdx.x & 15;                 // 16 col tiles of 64
  const int rowGroup = blockIdx.x >> 4;                 // 32 groups of 8 row tiles
  const int mbase = (rowGroup * 8 + wib) * 16;
  const int nbase = colTile * 64;

  const int sn = tid >> 2, sk = (tid & 3) * 8;          // staging split: 16B/thread
  auto stage = [&](int kb, int buf) {
    async_copy16(WT + (size_t)(nbase + sn) * EMB + kb + sk,
                 &bbufs[buf][sn * 32 + sk]);
  };

  const float* arow = X + (size_t)(mbase + l16) * EMB;
  float4 ar0, ar1, ar2, ar3;                            // raw next A fragment
  auto load_raw = [&](int kb) {
    const float* p1 = arow + kb + half * 16;
    const float* p2 = arow + kb + half * 16 + 8;
    ar0 = *(const float4*)p1; ar1 = *(const float4*)(p1 + 4);
    ar2 = *(const float4*)p2; ar3 = *(const float4*)(p2 + 4);
  };

  v8f c[4] = {};
  load_raw(0);
  stage(0, 0);

  for (int kb = 0; kb < EMB; kb += 32) {
    const int cur = (kb >> 5) & 1;
    v16bf a = cvt_a_from_f32(ar0, ar1, ar2, ar3);       // consumes last iter's loads
    if (kb + 32 < EMB) {
      load_raw(kb + 32);
      stage(kb + 32, cur ^ 1);
      wait_async_le(1);                                  // this tile's stage done
    } else {
      wait_async_le(0);
    }
    __syncthreads();
    const __bf16* bt = bbufs[cur];
#pragma unroll
    for (int t = 0; t < 4; ++t) {
      v16bf b = *(const v16bf*)(bt + (t * 16 + l16) * 32 + half * 16);
      c[t] = wmma_bf16(a, b, c[t]);
    }
    __syncthreads();
  }

#pragma unroll
  for (int t = 0; t < 4; ++t) {
    const int e = nbase + t * 16 + l16;                  // output feature
    const int h = e >> 6, d = e & (HD - 1);
#pragma unroll
    for (int r = 0; r < 8; ++r) {
      const int row = mbase + r + 8 * half;              // flattened b*S + s
      const int b = row >> 11, s = row & (S_LEN - 1);
      size_t idx;
      if (mode == 2)
        idx = (((size_t)(b * NH + h)) * HD + d) * S_LEN + s;   // v^T
      else
        idx = (((size_t)(b * NH + h)) * S_LEN + s) * HD + d;   // q, k
      out[idx] = (__bf16)c[t][r];
    }
  }
}

// ---- kernel 3: flash attention ---------------------------------------
// Block = 8 waves sharing one (b,h); each wave owns a 16-query tile.
// K/V tiles (32 keys) async-staged to LDS, double buffered.
__global__ __launch_bounds__(256) void attn_kernel(const __bf16* __restrict__ qbf,
                                                   const __bf16* __restrict__ kbf,
                                                   const __bf16* __restrict__ vT,
                                                   __bf16* __restrict__ ctx) {
  __shared__ __align__(32) __bf16 kbufs[2][32 * 64];    // [key][d]
  __shared__ __align__(32) __bf16 vbufs[2][64 * 32];    // [d][key]
  __shared__ __align__(32) __bf16 ldsP[8][16 * 32];     // per-wave P transpose

  const int tid  = threadIdx.x;
  const int lane = tid & 31, wib = tid >> 5;
  const int half = lane >> 4, l16 = lane & 15;
  const int bh = blockIdx.x >> 4;                        // 0..31 (b*NH+h)
  const int qt = (blockIdx.x & 15) * 8 + wib;            // query tile 0..127

  const __bf16* kg = kbf + (size_t)bh * S_LEN * HD;      // [S][64] (tiles contiguous)
  const __bf16* vg = vT  + (size_t)bh * HD * S_LEN;      // [64][S]
  __bf16* myP = ldsP[wib];

  const int vd = tid >> 2, vk = (tid & 3) * 8;           // V staging split
  auto stage = [&](int j, int buf) {
    async_copy16(kg + (size_t)j * 32 * 64 + tid * 8, &kbufs[buf][tid * 8]);
    async_copy16(vg + (size_t)vd * S_LEN + j * 32 + vk, &vbufs[buf][vd * 32 + vk]);
  };

  // Q fragments for the two K=32 steps over d=0..63 (kept in registers).
  const __bf16* qbase = qbf + ((size_t)bh * S_LEN + qt * 16 + l16) * HD;
  v16bf qf[2];
#pragma unroll
  for (int ks = 0; ks < 2; ++ks)
    qf[ks] = load_a_from_bf16(qbase + ks * 32 + half * 16,
                              qbase + ks * 32 + half * 16 + 8);

  v8f acc[4] = {};
  float mrun[8], lrun[8];
#pragma unroll
  for (int r = 0; r < 8; ++r) { mrun[r] = -1.0e30f; lrun[r] = 0.0f; }

  const float sc = 0.125f;                               // 1/sqrt(HD)
  stage(0, 0);

  for (int j = 0; j < S_LEN / 32; ++j) {
    const int cur = j & 1;
    if (j + 1 < S_LEN / 32) {
      stage(j + 1, cur ^ 1);
      wait_async_le(2);                                  // tile j's 2 copies done
    } else {
      wait_async_le(0);
    }
    __syncthreads();
    const __bf16* kt = kbufs[cur];
    const __bf16* vt = vbufs[cur];

    // scores: two 16x16 tiles (keys 0..15 / 16..31 of this block)
    v8f s0 = {}, s1 = {};
#pragma unroll
    for (int ks = 0; ks < 2; ++ks) {
      v16bf b0 = *(const v16bf*)(kt + (l16) * 64 + ks * 32 + half * 16);
      s0 = wmma_bf16(qf[ks], b0, s0);
      v16bf b1 = *(const v16bf*)(kt + (16 + l16) * 64 + ks * 32 + half * 16);
      s1 = wmma_bf16(qf[ks], b1, s1);
    }

    // online softmax per row (rows r+8*half live in this lane's 16-lane half)
    float p0[8], p1[8], alpha[8];
#pragma unroll
    for (int r = 0; r < 8; ++r) {
      float a0 = s0[r] * sc, a1 = s1[r] * sc;
      float mx = fmaxf(a0, a1);
      mx = fmaxf(mx, __shfl_xor(mx, 1));
      mx = fmaxf(mx, __shfl_xor(mx, 2));
      mx = fmaxf(mx, __shfl_xor(mx, 4));
      mx = fmaxf(mx, __shfl_xor(mx, 8));
      float mnew = fmaxf(mrun[r], mx);
      float al = __expf(mrun[r] - mnew);
      float e0 = __expf(a0 - mnew);
      float e1 = __expf(a1 - mnew);
      float rs = e0 + e1;
      rs += __shfl_xor(rs, 1);
      rs += __shfl_xor(rs, 2);
      rs += __shfl_xor(rs, 4);
      rs += __shfl_xor(rs, 8);
      lrun[r] = lrun[r] * al + rs;
      mrun[r] = mnew;
      alpha[r] = al; p0[r] = e0; p1[r] = e1;
    }

#pragma unroll
    for (int t = 0; t < 4; ++t)
#pragma unroll
      for (int r = 0; r < 8; ++r) acc[t][r] *= alpha[r];

    // transpose P (C/D layout -> A layout) through LDS, as bf16
#pragma unroll
    for (int r = 0; r < 8; ++r) {
      const int m = r + 8 * half;
      myP[m * 32 + l16]      = (__bf16)p0[r];
      myP[m * 32 + 16 + l16] = (__bf16)p1[r];
    }
    asm volatile("s_wait_dscnt 0" ::: "memory");         // same-wave DS in-order
    const __bf16* pr = myP + l16 * 32;
    v16bf pf = load_a_from_bf16(pr + half * 16, pr + half * 16 + 8);

    // acc += P @ V  (K = 32 keys, 4 d-tiles of 16)
#pragma unroll
    for (int t = 0; t < 4; ++t) {
      v16bf bv = *(const v16bf*)(vt + (t * 16 + l16) * 32 + half * 16);
      acc[t] = wmma_bf16(pf, bv, acc[t]);
    }
    __syncthreads();                                     // done reading buf[cur]
  }

  // epilogue: divide by row sums, write bf16 context [B,S,E]
  const int b = bh >> 4, h = bh & 15;
#pragma unroll
  for (int t = 0; t < 4; ++t) {
    const int d = t * 16 + l16;
#pragma unroll
    for (int r = 0; r < 8; ++r) {
      const int s = qt * 16 + r + 8 * half;
      const float inv = 1.0f / lrun[r];
      ctx[((size_t)b * S_LEN + s) * EMB + h * HD + d] = (__bf16)(acc[t][r] * inv);
    }
  }
}

// ---- kernel 4: output projection  out = ctx @ W_o  (f32 result) -------
__global__ __launch_bounds__(256) void proj_out_kernel(const __bf16* __restrict__ A,
                                                       const __bf16* __restrict__ WT,
                                                       float* __restrict__ out) {
  __shared__ __align__(32) __bf16 bbufs[2][64 * 32];

  const int tid  = threadIdx.x;
  const int lane = tid & 31, wib = tid >> 5;
  const int half = lane >> 4, l16 = lane & 15;
  const int colTile  = blockIdx.x & 15;
  const int rowGroup = blockIdx.x >> 4;
  const int mbase = (rowGroup * 8 + wib) * 16;
  const int nbase = colTile * 64;

  const int sn = tid >> 2, sk = (tid & 3) * 8;
  auto stage = [&](int kb, int buf) {
    async_copy16(WT + (size_t)(nbase + sn) * EMB + kb + sk,
                 &bbufs[buf][sn * 32 + sk]);
  };

  const __bf16* arow = A + (size_t)(mbase + l16) * EMB;
  v8bf r1, r2;
  auto load_raw = [&](int kb) {
    r1 = *(const v8bf*)(arow + kb + half * 16);
    r2 = *(const v8bf*)(arow + kb + half * 16 + 8);
  };

  v8f c[4] = {};
  load_raw(0);
  stage(0, 0);

  for (int kb = 0; kb < EMB; kb += 32) {
    const int cur = (kb >> 5) & 1;
    v16bf a;
#pragma unroll
    for (int j = 0; j < 8; ++j) { a[j] = r1[j]; a[j + 8] = r2[j]; }
    if (kb + 32 < EMB) {
      load_raw(kb + 32);
      stage(kb + 32, cur ^ 1);
      wait_async_le(1);
    } else {
      wait_async_le(0);
    }
    __syncthreads();
    const __bf16* bt = bbufs[cur];
#pragma unroll
    for (int t = 0; t < 4; ++t) {
      v16bf b = *(const v16bf*)(bt + (t * 16 + l16) * 32 + half * 16);
      c[t] = wmma_bf16(a, b, c[t]);
    }
    __syncthreads();
  }

#pragma unroll
  for (int t = 0; t < 4; ++t) {
    const int col = nbase + t * 16 + l16;
#pragma unroll
    for (int r = 0; r < 8; ++r) {
      const int row = mbase + r + 8 * half;
      out[(size_t)row * EMB + col] = c[t][r];
    }
  }
}

// ---- host-side launcher ----------------------------------------------
extern "C" void kernel_launch(void* const* d_in, const int* in_sizes, int n_in,
                              void* d_out, int out_size, void* d_ws, size_t ws_size,
                              hipStream_t stream) {
  (void)in_sizes; (void)n_in; (void)out_size; (void)ws_size;

  const float* Q  = (const float*)d_in[0];
  const float* K  = (const float*)d_in[1];
  const float* V  = (const float*)d_in[2];
  const float* Wq = (const float*)d_in[3];
  const float* Wk = (const float*)d_in[4];
  const float* Wv = (const float*)d_in[5];
  const float* Wo = (const float*)d_in[6];
  float* out = (float*)d_out;

  // workspace layout (bf16 elements): 4 transposed weights + q + k + v^T + ctx
  __bf16* ws  = (__bf16*)d_ws;
  __bf16* WqT = ws;
  __bf16* WkT = WqT + (size_t)EMB * EMB;
  __bf16* WvT = WkT + (size_t)EMB * EMB;
  __bf16* WoT = WvT + (size_t)EMB * EMB;
  __bf16* qbf = WoT + (size_t)EMB * EMB;
  __bf16* kbf = qbf + (size_t)MROWS * EMB;
  __bf16* vTb = kbf + (size_t)MROWS * EMB;
  __bf16* ctx = vTb + (size_t)MROWS * EMB;   // total 20M bf16 = 40 MB

  const dim3 blk(256);
  const int wt_blocks   = (EMB * EMB) / 256;                       // 4096
  const int gemm_blocks = ((MROWS / 16) / 8) * (EMB / 64);         // 512
  const int attn_blocks = BATCH * NH * (S_LEN / 16 / 8);           // 512

  wtrans_kernel<<<wt_blocks, blk, 0, stream>>>(Wq, WqT);
  wtrans_kernel<<<wt_blocks, blk, 0, stream>>>(Wk, WkT);
  wtrans_kernel<<<wt_blocks, blk, 0, stream>>>(Wv, WvT);
  wtrans_kernel<<<wt_blocks, blk, 0, stream>>>(Wo, WoT);

  proj_in_kernel<<<gemm_blocks, blk, 0, stream>>>(Q, WqT, qbf, 0);
  proj_in_kernel<<<gemm_blocks, blk, 0, stream>>>(K, WkT, kbf, 0);
  proj_in_kernel<<<gemm_blocks, blk, 0, stream>>>(V, WvT, vTb, 2);

  attn_kernel<<<attn_blocks, blk, 0, stream>>>(qbf, kbf, vTb, ctx);

  proj_out_kernel<<<gemm_blocks, blk, 0, stream>>>(ctx, WoT, out);
}
